// UPST_DirectVoxGO_80212809220442
// MI455X (gfx1250) — matrix operand order repaired and checked by reading, not compiled, not run on hardware
//
#include <hip/hip_runtime.h>
#include <hip/hip_bf16.h>
#include <stdint.h>

typedef _Float16 h16v  __attribute__((ext_vector_type(16)));
typedef float    f32x8 __attribute__((ext_vector_type(8)));

namespace {
constexpr int RAYS = 4096;
constexpr int SAMP = 256;
constexpr int G    = 160;
constexpr long long G3 = (long long)G * G * G;
constexpr int K0D  = 12;
constexpr int WID  = 128;
constexpr float TNEAR = 0.2f, TFAR = 1.8f;
constexpr float ACT_SHIFT = -4.5951198501345898f; // log(1/(1-0.01) - 1)

// workspace byte offsets
constexpr size_t OFF_VW   = 0;                                   // [4096][128] f32
constexpr size_t OFF_WT   = OFF_VW + (size_t)RAYS * WID * 4;     // [4096][256] f32 (alpha -> weights)
constexpr size_t OFF_FEAT = OFF_WT + (size_t)RAYS * SAMP * 4;    // [1M][16] f16

// LDS layout for the WMMA MLP kernel (half-element indices)
constexpr int W0A_STRIDE = 36;   // W0aT: [128][36], K=0..31 used (rows>=12 zero)
constexpr int W1_STRIDE  = 132;  // W1T:  [128][132]
constexpr int W2_STRIDE  = 132;  // W2T:  [16][132]  (cols n>=3 zero)
constexpr int ST_STRIDE  = 136;  // staging per wave: [16][136]
constexpr int H_W0A = 0;
constexpr int H_W1  = H_W0A + 128 * W0A_STRIDE;
constexpr int H_W2  = H_W1  + 128 * W1_STRIDE;
constexpr int H_ST  = H_W2  + 16  * W2_STRIDE;
constexpr int H_END = H_ST  + 8 * 16 * ST_STRIDE;
constexpr int F_OFF = H_END * 2;                 // byte offset of f32 region
constexpr int N_FLT = 128 + 16 + 128 + 64;       // b1s, b2s, vWs, red
constexpr size_t SMEM_BYTES = (size_t)F_OFF + (size_t)N_FLT * 4;

union Af { uint32_t u[8]; h16v h; };
} // namespace

// ---------------------------------------------------------------------------
// Kernel 1: per-ray view embedding folded into W0 tail: vW[r][n] = b0[n] + vemb@W0[12:]
// ---------------------------------------------------------------------------
__global__ __launch_bounds__(128) void k_ray_embed(
    const float* __restrict__ rays_d, const float* __restrict__ W0,
    const float* __restrict__ b0, float* __restrict__ vW)
{
  int r = blockIdx.x, n = threadIdx.x;
  float dx = rays_d[r*3+0], dy = rays_d[r*3+1], dz = rays_d[r*3+2];
  float inv = rsqrtf(dx*dx + dy*dy + dz*dz);
  float d[3] = {dx*inv, dy*inv, dz*inv};
  float acc = b0[n];
  #pragma unroll
  for (int a = 0; a < 3; a++) acc += d[a] * W0[(12 + a) * WID + n];
  #pragma unroll
  for (int a = 0; a < 3; a++) {
    #pragma unroll
    for (int p = 0; p < 4; p++) {
      float ang = d[a] * (float)(1 << p);
      acc += sinf(ang) * W0[(15 + a*4 + p) * WID + n];
      acc += cosf(ang) * W0[(27 + a*4 + p) * WID + n];
    }
  }
  vW[(size_t)r * WID + n] = acc;
}

// ---------------------------------------------------------------------------
// Kernel 2: per-sample trilinear of density (-> alpha) and 12-ch k0 (-> f16 feat)
// ---------------------------------------------------------------------------
__global__ __launch_bounds__(256) void k_sample(
    const float* __restrict__ rays_o, const float* __restrict__ rays_d,
    const float* __restrict__ density, const float* __restrict__ k0,
    float* __restrict__ alpha_out, _Float16* __restrict__ feat16)
{
  int s  = blockIdx.x * 256 + threadIdx.x;
  int r  = s >> 8, si = s & 255;
  float t = TNEAR + (TFAR - TNEAR) * ((float)si * (1.0f / 255.0f));
  float p[3];
  #pragma unroll
  for (int a = 0; a < 3; a++) p[a] = rays_o[r*3+a] + rays_d[r*3+a] * t;

  int lo[3]; float f[3];
  #pragma unroll
  for (int a = 0; a < 3; a++) {
    float g = (p[a] + 1.0f) * 0.5f * (float)(G - 1);
    g = fminf(fmaxf(g, 0.0f), (float)(G - 1));
    int l = (int)floorf(g);
    l = min(l, G - 2);
    lo[a] = l;
    f[a]  = g - (float)l;
  }
  size_t base = ((size_t)lo[0] * G + lo[1]) * G + lo[2];
  const size_t dX = (size_t)G * G, dY = G, dZ = 1;
  float wx1=f[0], wx0=1.f-wx1, wy1=f[1], wy0=1.f-wy1, wz1=f[2], wz0=1.f-wz1;
  float w000=wx0*wy0*wz0, w001=wx0*wy0*wz1, w010=wx0*wy1*wz0, w011=wx0*wy1*wz1;
  float w100=wx1*wy0*wz0, w101=wx1*wy0*wz1, w110=wx1*wy1*wz0, w111=wx1*wy1*wz1;

  const float* g0 = density + base;
  float sigma = w000*g0[0]     + w001*g0[dZ]     + w010*g0[dY]     + w011*g0[dY+dZ]
              + w100*g0[dX]    + w101*g0[dX+dZ]  + w110*g0[dX+dY]  + w111*g0[dX+dY+dZ];
  float alpha = 1.0f - rsqrtf(1.0f + __expf(sigma + ACT_SHIFT)); // (1+e)^-0.5
  alpha_out[s] = alpha;

  _Float16* fr = feat16 + (size_t)s * 16;
  #pragma unroll
  for (int c = 0; c < K0D; c++) {
    const float* gc = k0 + (size_t)c * G3 + base;
    float v = w000*gc[0]    + w001*gc[dZ]    + w010*gc[dY]    + w011*gc[dY+dZ]
            + w100*gc[dX]   + w101*gc[dX+dZ] + w110*gc[dX+dY] + w111*gc[dX+dY+dZ];
    fr[c] = (_Float16)v;
  }
  #pragma unroll
  for (int c = K0D; c < 16; c++) fr[c] = (_Float16)0.0f;
}

// ---------------------------------------------------------------------------
// Kernel 3: per-ray cumprod scan -> weights; seed d_out with alphainv_last
// ---------------------------------------------------------------------------
__global__ __launch_bounds__(256) void k_scan(
    float* __restrict__ wbuf, float* __restrict__ out)
{
  __shared__ float sc[256];
  int r = blockIdx.x, s = threadIdx.x;
  float a = wbuf[(size_t)r * SAMP + s];
  sc[s] = 1.0f - a + 1e-10f;
  __syncthreads();
  #pragma unroll
  for (int off = 1; off < 256; off <<= 1) {
    float v = (s >= off) ? sc[s - off] : 1.0f;
    __syncthreads();
    sc[s] *= v;
    __syncthreads();
  }
  float T = (s == 0) ? 1.0f : sc[s - 1];
  wbuf[(size_t)r * SAMP + s] = a * T;
  if (s < 3) out[r * 3 + s] = sc[255]; // alphainv_last seed
}

// ---------------------------------------------------------------------------
// Kernel 4: fused WMMA MLP + weighted ray accumulation. One block per ray.
// 8 waves x 2 iterations, each wave-tile = 16 samples (M=16).
// ---------------------------------------------------------------------------
__global__ __launch_bounds__(256) void k_mlp(
    const float* __restrict__ W0, const float* __restrict__ W1,
    const float* __restrict__ b1, const float* __restrict__ W2,
    const float* __restrict__ b2, const float* __restrict__ vW,
    const float* __restrict__ weights, const _Float16* __restrict__ feat16,
    float* __restrict__ out)
{
  extern __shared__ char smem[];
  _Float16* lds = (_Float16*)smem;
  float* fl  = (float*)(smem + F_OFF);
  float* b1s = fl;             // 128
  float* b2s = fl + 128;       // 16
  float* vWs = fl + 144;       // 128
  float* red = fl + 272;       // 8 waves * 2 halves * 4 (3 used)

  int tid = threadIdx.x;
  int r   = blockIdx.x;

  // Stage transposed f16 weights into LDS ([N][K] so B-fragments are contiguous dwords)
  for (int idx = tid; idx < 128 * W0A_STRIDE; idx += 256) {
    int n = idx / W0A_STRIDE, k = idx % W0A_STRIDE;
    lds[H_W0A + idx] = (k < 12) ? (_Float16)W0[k * WID + n] : (_Float16)0.0f;
  }
  for (int idx = tid; idx < 128 * W1_STRIDE; idx += 256) {
    int n = idx / W1_STRIDE, k = idx % W1_STRIDE;
    lds[H_W1 + idx] = (k < 128) ? (_Float16)W1[k * WID + n] : (_Float16)0.0f;
  }
  for (int idx = tid; idx < 16 * W2_STRIDE; idx += 256) {
    int n = idx / W2_STRIDE, k = idx % W2_STRIDE;
    lds[H_W2 + idx] = (k < 128 && n < 3) ? (_Float16)W2[k * 3 + n] : (_Float16)0.0f;
  }
  if (tid < 128) b1s[tid] = b1[tid];
  if (tid < 16)  b2s[tid] = (tid < 3) ? b2[tid] : 0.0f;
  if (tid < 128) vWs[tid] = vW[(size_t)r * WID + tid];
  if (tid < 64)  red[tid] = 0.0f;
  __syncthreads();

  int wave = tid >> 5, lane = tid & 31;
  int m = lane & 15, hi = lane >> 4;      // A: row m; K halves split by hi
  _Float16* st = lds + H_ST + wave * 16 * ST_STRIDE;

  for (int iter = 0; iter < 2; iter++) {
    int tile = wave + iter * 8;
    int sb = r * SAMP + tile * 16;        // first sample of this 16-row tile

    // ---- A1 (feat, 16x32, K=0..11 real, rest zero) ----
    Af a1;
    const uint32_t* fp = (const uint32_t*)(feat16 + (size_t)(sb + m) * 16);
    #pragma unroll
    for (int v = 0; v < 8; v++) a1.u[v] = hi ? 0u : fp[v];  // K 0..15 / 16..31 (zero)

    // ---- GEMM1: h0 = relu(feat @ W0a + vW[r]) ----
    #pragma unroll
    for (int nt = 0; nt < 8; nt++) {
      int n = nt * 16 + m;
      float cv = vWs[n];
      f32x8 c;
      #pragma unroll
      for (int v = 0; v < 8; v++) c[v] = cv;
      Af b;
      const uint32_t* bp = (const uint32_t*)(lds + H_W0A + n * W0A_STRIDE + hi * 16);
      #pragma unroll
      for (int v = 0; v < 8; v++) b.u[v] = bp[v];
      f32x8 d = __builtin_amdgcn_wmma_f32_16x16x32_f16(false, a1.h, false, b.h,
                                                       (short)0, c, false, false);
      #pragma unroll
      for (int v = 0; v < 8; v++)
        st[(v + hi * 8) * ST_STRIDE + n] = (_Float16)fmaxf(d[v], 0.0f);
    }

    // ---- A2 from staging (D->A layout transpose via LDS) ----
    Af a2[4];
    #pragma unroll
    for (int kc = 0; kc < 4; kc++) {
      int base = m * ST_STRIDE + kc * 32 + hi * 16;
      #pragma unroll
      for (int v = 0; v < 4; v++) a2[kc].u[v] = *(const uint32_t*)(st + base + 2 * v);
      #pragma unroll
      for (int v = 4; v < 8; v++) a2[kc].u[v] = *(const uint32_t*)(st + base + 8 + 2 * (v - 4));
    }

    // ---- GEMM2: h1 = relu(h0 @ W1 + b1), 8 N-tiles x 4 K-chunks ----
    f32x8 h1acc[8];
    #pragma unroll
    for (int nt = 0; nt < 8; nt++) {
      int n = nt * 16 + m;
      float cv = b1s[n];
      f32x8 c;
      #pragma unroll
      for (int v = 0; v < 8; v++) c[v] = cv;
      #pragma unroll
      for (int kc = 0; kc < 4; kc++) {
        Af b;
        const uint32_t* bp = (const uint32_t*)(lds + H_W1 + n * W1_STRIDE + kc * 32 + hi * 16);
        #pragma unroll
        for (int v = 0; v < 8; v++) b.u[v] = bp[v];
        c = __builtin_amdgcn_wmma_f32_16x16x32_f16(false, a2[kc].h, false, b.h,
                                                   (short)0, c, false, false);
      }
      h1acc[nt] = c;
    }
    #pragma unroll
    for (int nt = 0; nt < 8; nt++) {
      #pragma unroll
      for (int v = 0; v < 8; v++)
        st[(v + hi * 8) * ST_STRIDE + nt * 16 + m] = (_Float16)fmaxf(h1acc[nt][v], 0.0f);
    }

    // ---- A3 from staging ----
    Af a3[4];
    #pragma unroll
    for (int kc = 0; kc < 4; kc++) {
      int base = m * ST_STRIDE + kc * 32 + hi * 16;
      #pragma unroll
      for (int v = 0; v < 4; v++) a3[kc].u[v] = *(const uint32_t*)(st + base + 2 * v);
      #pragma unroll
      for (int v = 4; v < 8; v++) a3[kc].u[v] = *(const uint32_t*)(st + base + 8 + 2 * (v - 4));
    }

    // ---- GEMM3: rgb = sigmoid(h1 @ W2 + b2), single padded N-tile ----
    f32x8 c3;
    #pragma unroll
    for (int v = 0; v < 8; v++) c3[v] = b2s[m];
    #pragma unroll
    for (int kc = 0; kc < 4; kc++) {
      Af b;
      const uint32_t* bp = (const uint32_t*)(lds + H_W2 + m * W2_STRIDE + kc * 32 + hi * 16);
      #pragma unroll
      for (int v = 0; v < 8; v++) b.u[v] = bp[v];
      c3 = __builtin_amdgcn_wmma_f32_16x16x32_f16(false, a3[kc].h, false, b.h,
                                                  (short)0, c3, false, false);
    }

    // ---- weighted accumulation: sum_s weights[s] * sigmoid(rgb[s][n]) ----
    float psum = 0.0f;
    #pragma unroll
    for (int v = 0; v < 8; v++) {
      int row = v + hi * 8;                       // sample row in tile
      float wgt  = weights[(size_t)sb + row];
      float rgbv = 1.0f / (1.0f + __expf(-c3[v]));
      psum += wgt * rgbv;
    }
    if (m < 3) red[(wave * 2 + hi) * 4 + m] += psum;  // per-thread slot, both iters
  }
  __syncthreads();
  if (tid < 3) {
    float acc = out[r * 3 + tid];      // alphainv_last seeded by k_scan
    #pragma unroll
    for (int i = 0; i < 16; i++) acc += red[i * 4 + tid];
    out[r * 3 + tid] = acc;
  }
}

// ---------------------------------------------------------------------------
extern "C" void kernel_launch(void* const* d_in, const int* in_sizes, int n_in,
                              void* d_out, int out_size, void* d_ws, size_t ws_size,
                              hipStream_t stream)
{
  const float* rays_o  = (const float*)d_in[0];
  const float* rays_d  = (const float*)d_in[1];
  const float* density = (const float*)d_in[2];
  const float* k0      = (const float*)d_in[3];
  const float* W0      = (const float*)d_in[4];
  const float* b0      = (const float*)d_in[5];
  const float* W1      = (const float*)d_in[6];
  const float* b1      = (const float*)d_in[7];
  const float* W2      = (const float*)d_in[8];
  const float* b2      = (const float*)d_in[9];
  float* out = (float*)d_out;
  char*  ws  = (char*)d_ws;

  float*    vW     = (float*)(ws + OFF_VW);
  float*    wbuf   = (float*)(ws + OFF_WT);
  _Float16* feat16 = (_Float16*)(ws + OFF_FEAT);

  k_ray_embed<<<RAYS, 128, 0, stream>>>(rays_d, W0, b0, vW);
  k_sample<<<(RAYS * SAMP) / 256, 256, 0, stream>>>(rays_o, rays_d, density, k0, wbuf, feat16);
  k_scan<<<RAYS, 256, 0, stream>>>(wbuf, out);
  k_mlp<<<RAYS, 256, SMEM_BYTES, stream>>>(W0, W1, b1, W2, b2, vW, wbuf, feat16, out);
}
